// CasualSelfAttention_66786741453344
// MI455X (gfx1250) — matrix operand order repaired and checked by reading, not compiled, not run on hardware
//
#include <hip/hip_runtime.h>
#include <hip/hip_bf16.h>
#include <cstdint>

typedef __bf16 bf16x16_t __attribute__((ext_vector_type(16)));
typedef float  f32x8_t  __attribute__((ext_vector_type(8)));
typedef int    v4i_t    __attribute__((vector_size(16)));

#define WMMA_BF16(a, b, c) \
  __builtin_amdgcn_wmma_f32_16x16x32_bf16(false, (a), false, (b), (short)0, (c), false, false)

union FragB16 { bf16x16_t v; unsigned int u[8]; };

#if defined(__HIP_DEVICE_COMPILE__) && __has_builtin(__builtin_amdgcn_global_load_async_to_lds_b128)
#define ASYNC_LDS 1
#endif

__device__ __forceinline__ void async_copy_b128(const void* g, void* l) {
#ifdef ASYNC_LDS
  __builtin_amdgcn_global_load_async_to_lds_b128(
      (__attribute__((address_space(1))) v4i_t*)g,
      (__attribute__((address_space(3))) v4i_t*)l, 0, 0);
#else
  *(uint4*)l = *(const uint4*)g;
#endif
}

__device__ __forceinline__ void wait_async() {
#ifdef ASYNC_LDS
#if __has_builtin(__builtin_amdgcn_s_wait_asynccnt)
  __builtin_amdgcn_s_wait_asynccnt(0);
#else
  asm volatile("s_wait_asynccnt 0" ::: "memory");
#endif
#endif
}

__device__ __forceinline__ unsigned short f32_to_bf16(float f) {
  union { float f; unsigned int u; } x; x.f = f;
  unsigned int u = x.u;
  unsigned int r = u + 0x7FFFu + ((u >> 16) & 1u);   // round-to-nearest-even
  return (unsigned short)(r >> 16);
}

// A-fragment dword offset (within a padded 20-dword K=32 row) for element v,
// contiguous k order (k = 16*hi + 2v), matching B fragments. hi = lane>>4.
__device__ __forceinline__ int a_dw(int v, int hi) {
  return hi * 8 + v;
}

// ---------------------------------------------------------------- convert
__global__ void cvt_f32_bf16(const float* __restrict__ src,
                             unsigned short* __restrict__ dst, int n) {
  int i = (blockIdx.x * blockDim.x + threadIdx.x) * 4;
  if (i + 3 < n) {
    float4 f = *(const float4*)(src + i);
    dst[i + 0] = f32_to_bf16(f.x);
    dst[i + 1] = f32_to_bf16(f.y);
    dst[i + 2] = f32_to_bf16(f.z);
    dst[i + 3] = f32_to_bf16(f.w);
  }
}

// ---------------------------------------------------------------- shared GEMM core
// A [8192][1024] bf16 row-major, W [1024][LDW] bf16 row-major.
// Block tile 128x128, K-step 32, double-buffered LDS, async A staging.
// ldsA/ldsB are [2][2560] dword buffers (rows padded to 20 dwords).
template <int LDW>
__device__ __forceinline__ void gemm_mma(const unsigned short* __restrict__ Ab,
                                         const unsigned short* __restrict__ Wb,
                                         unsigned int* ldsA, unsigned int* ldsB,
                                         int m0, int n0, f32x8_t (&acc)[8]) {
  const int tid = threadIdx.x;
  const int wave = tid >> 5, lane = tid & 31;
  const int laneN = lane & 15, hi = lane >> 4;

  // ---- prologue: stage tile kk=0 into buffer 0
  {
    uint4 breg[2];
#pragma unroll
    for (int i = 0; i < 2; ++i) {
      int c = tid + i * 256;
      int row = c >> 2, col = c & 3;
      async_copy_b128(Ab + (m0 + row) * 1024 + col * 8, ldsA + row * 20 + col * 4);
      int krow = c >> 4, cg = c & 15;
      breg[i] = *(const uint4*)(Wb + krow * LDW + n0 + cg * 8);
    }
    unsigned short* bs = (unsigned short*)ldsB;
#pragma unroll
    for (int i = 0; i < 2; ++i) {
      int c = tid + i * 256;
      int krow = c >> 4, nb = (c & 15) * 8;
      const unsigned short* ds = (const unsigned short*)&breg[i];
#pragma unroll
      for (int j = 0; j < 8; ++j) bs[(nb + j) * 40 + krow] = ds[j];  // transpose
    }
    wait_async();
  }
  __syncthreads();

  int buf = 0;
  for (int kk = 0; kk < 1024; kk += 32) {
    unsigned int* curA = ldsA + buf * 2560;
    unsigned int* curB = ldsB + buf * 2560;
    unsigned int* nxtA = ldsA + (buf ^ 1) * 2560;
    unsigned int* nxtB = ldsB + (buf ^ 1) * 2560;
    const int nk = kk + 32;

    // ---- issue next tile's global traffic early (latency hidden by WMMAs)
    uint4 breg[2];
    if (nk < 1024) {
#pragma unroll
      for (int i = 0; i < 2; ++i) {
        int c = tid + i * 256;
        int row = c >> 2, col = c & 3;
        async_copy_b128(Ab + (m0 + row) * 1024 + nk + col * 8,
                        nxtA + row * 20 + col * 4);
        int krow = c >> 4, cg = c & 15;
        breg[i] = *(const uint4*)(Wb + (nk + krow) * LDW + n0 + cg * 8);
      }
    }

    // ---- compute: batch all fragment loads, then back-to-back WMMAs
    FragB16 fa, fb[8];
#pragma unroll
    for (int v = 0; v < 8; ++v)
      fa.u[v] = curA[(wave * 16 + laneN) * 20 + a_dw(v, hi)];
#pragma unroll
    for (int t = 0; t < 8; ++t)
#pragma unroll
      for (int v = 0; v < 8; ++v)
        fb[t].u[v] = curB[(t * 16 + laneN) * 20 + hi * 8 + v];
#pragma unroll
    for (int t = 0; t < 8; ++t)
      acc[t] = WMMA_BF16(fa.v, fb[t].v, acc[t]);

    // ---- finish staging next B tile (transpose scatter), sync
    if (nk < 1024) {
      unsigned short* bs = (unsigned short*)nxtB;
#pragma unroll
      for (int i = 0; i < 2; ++i) {
        int c = tid + i * 256;
        int krow = c >> 4, nb = (c & 15) * 8;
        const unsigned short* ds = (const unsigned short*)&breg[i];
#pragma unroll
        for (int j = 0; j < 8; ++j) bs[(nb + j) * 40 + krow] = ds[j];
      }
      wait_async();
    }
    __syncthreads();
    buf ^= 1;
  }
}

// ---------------------------------------------------------------- GEMM1: qkv
__global__ __launch_bounds__(256)
void gemm_qkv(const unsigned short* __restrict__ Xb,
              const unsigned short* __restrict__ Wb,
              const float* __restrict__ bias,
              unsigned short* __restrict__ Qo,
              unsigned short* __restrict__ Ko,
              unsigned short* __restrict__ Vo) {
  __shared__ __align__(16) unsigned int ldsA[2 * 2560];
  __shared__ __align__(16) unsigned int ldsB[2 * 2560];
  const int tid = threadIdx.x;
  const int wave = tid >> 5, lane = tid & 31;
  const int laneN = lane & 15, hi = lane >> 4;
  const int m0 = (blockIdx.x & 63) * 128;
  const int n0 = (blockIdx.x >> 6) * 128;     // 24 N-tiles

  f32x8_t acc[8];
  for (int t = 0; t < 8; ++t)
    for (int i = 0; i < 8; ++i) acc[t][i] = 0.f;

  gemm_mma<3072>(Xb, Wb, ldsA, ldsB, m0, n0, acc);

  // bias + scatter to q/k/v in [B=4,H=16,T=2048,hd=64]
#pragma unroll
  for (int t = 0; t < 8; ++t) {
    int n = n0 + t * 16 + laneN;
    float bv = bias[n];
    int sec = n >> 10;
    int c = n & 1023;
    int h = c >> 6, d = c & 63;
    unsigned short* dst = (sec == 0) ? Qo : ((sec == 1) ? Ko : Vo);
#pragma unroll
    for (int r = 0; r < 8; ++r) {
      int m = m0 + wave * 16 + hi * 8 + r;
      int b = m >> 11, tt = m & 2047;
      dst[(((b * 16 + h) * 2048) + tt) * 64 + d] = f32_to_bf16(acc[t][r] + bv);
    }
  }
}

// ---------------------------------------------------------------- GEMM2: out proj
__global__ __launch_bounds__(256)
void gemm_proj(const unsigned short* __restrict__ Yb,
               const unsigned short* __restrict__ Wb,
               const float* __restrict__ bias,
               float* __restrict__ Out) {
  __shared__ __align__(16) unsigned int ldsA[2 * 2560];
  __shared__ __align__(16) unsigned int ldsB[2 * 2560];
  const int tid = threadIdx.x;
  const int wave = tid >> 5, lane = tid & 31;
  const int laneN = lane & 15, hi = lane >> 4;
  const int m0 = (blockIdx.x & 63) * 128;
  const int n0 = (blockIdx.x >> 6) * 128;     // 8 N-tiles

  f32x8_t acc[8];
  for (int t = 0; t < 8; ++t)
    for (int i = 0; i < 8; ++i) acc[t][i] = 0.f;

  gemm_mma<1024>(Yb, Wb, ldsA, ldsB, m0, n0, acc);

#pragma unroll
  for (int t = 0; t < 8; ++t) {
    int n = n0 + t * 16 + laneN;
    float bv = bias[n];
#pragma unroll
    for (int r = 0; r < 8; ++r) {
      int m = m0 + wave * 16 + hi * 8 + r;
      Out[(size_t)m * 1024 + n] = acc[t][r] + bv;
    }
  }
}

// ---------------------------------------------------------------- flash attention
// Q,K,V bf16 [B,H,T,64] -> Y bf16 [B,T,C]. Double-buffered K/V tiles; K async.
__global__ __launch_bounds__(256)
void flash_attn(const unsigned short* __restrict__ Qb,
                const unsigned short* __restrict__ Kb,
                const unsigned short* __restrict__ Vb,
                unsigned short* __restrict__ Yb) {
  __shared__ __align__(16) unsigned int ldsK[2 * 32 * 36];   // [32 key][64 d], pad 36dw
  __shared__ __align__(16) unsigned int ldsV[2 * 64 * 20];   // Vt [64 d][32 key], pad 20dw
  __shared__ __align__(16) unsigned int ldsP[8 * 16 * 20];   // per-wave [16 q][32 key]
  const int tid = threadIdx.x;
  const int wave = tid >> 5, lane = tid & 31;
  const int laneN = lane & 15, hi = lane >> 4;
  const int qt = blockIdx.x & 15;
  const int h  = (blockIdx.x >> 4) & 15;
  const int b  = blockIdx.x >> 8;
  const int q0 = qt * 128;
  const unsigned short* Q = Qb + (size_t)((b * 16 + h) * 2048) * 64;
  const unsigned short* K = Kb + (size_t)((b * 16 + h) * 2048) * 64;
  const unsigned short* V = Vb + (size_t)((b * 16 + h) * 2048) * 64;

  const int row = tid >> 3, cg = tid & 7;   // 256 chunks: 32 rows x 8 groups of 8 bf16

  // Q fragments for this wave's 16 rows (resident in VGPRs for whole kernel)
  FragB16 qf[2];
  {
    int qrow = q0 + wave * 16 + laneN;
    const unsigned int* qp = (const unsigned int*)(Q + qrow * 64);
#pragma unroll
    for (int f = 0; f < 2; ++f)
#pragma unroll
      for (int v = 0; v < 8; ++v) qf[f].u[v] = qp[f * 16 + a_dw(v, hi)];
  }

  f32x8_t o[4];
  for (int t = 0; t < 4; ++t)
    for (int i = 0; i < 8; ++i) o[t][i] = 0.f;
  float mrow[8], lrow[8];
  for (int r = 0; r < 8; ++r) { mrow[r] = -__builtin_inff(); lrow[r] = 0.f; }

  const int nkt = (q0 + 128) >> 5;          // causal: key tiles up to diagonal

  // ---- prologue: stage key-tile 0 into buffer 0
  {
    async_copy_b128(K + row * 64 + cg * 8, ldsK + row * 36 + cg * 4);
    uint4 d4 = *(const uint4*)(V + row * 64 + cg * 8);
    unsigned short* vs = (unsigned short*)ldsV;
    const unsigned short* ds = (const unsigned short*)&d4;
#pragma unroll
    for (int j = 0; j < 8; ++j) vs[(cg * 8 + j) * 40 + row] = ds[j];   // transpose
    wait_async();
  }
  __syncthreads();

  int buf = 0;
  for (int kt = 0; kt < nkt; ++kt) {
    const int kbase = kt << 5;
    unsigned int* curK = ldsK + buf * (32 * 36);
    unsigned int* curV = ldsV + buf * (64 * 20);
    unsigned int* nxtK = ldsK + (buf ^ 1) * (32 * 36);
    unsigned int* nxtV = ldsV + (buf ^ 1) * (64 * 20);
    const bool more = (kt + 1) < nkt;

    // issue next tile's global traffic
    uint4 vreg;
    if (more) {
      async_copy_b128(K + (kbase + 32 + row) * 64 + cg * 8, nxtK + row * 36 + cg * 4);
      vreg = *(const uint4*)(V + (kbase + 32 + row) * 64 + cg * 8);
    }

    // S = Q * K^T : 16 queries x 32 keys (two 16x16 tiles, 2 WMMA each)
    f32x8_t s[2];
#pragma unroll
    for (int st = 0; st < 2; ++st) {
      for (int i = 0; i < 8; ++i) s[st][i] = 0.f;
      FragB16 kf[2];
#pragma unroll
      for (int f = 0; f < 2; ++f)
#pragma unroll
        for (int v = 0; v < 8; ++v)
          kf[f].u[v] = curK[(st * 16 + laneN) * 36 + f * 16 + hi * 8 + v];
#pragma unroll
      for (int f = 0; f < 2; ++f)
        s[st] = WMMA_BF16(qf[f].v, kf[f].v, s[st]);
    }

    // online softmax; row = hi*8+r spans the 16 lanes of this half-wave
    unsigned short* pp = (unsigned short*)(ldsP + wave * 320);
#pragma unroll
    for (int r = 0; r < 8; ++r) {
      int qidx = q0 + wave * 16 + hi * 8 + r;
      float s0 = s[0][r] * 0.125f;            // 1/sqrt(64)
      float s1 = s[1][r] * 0.125f;
      if (kbase + laneN > qidx)      s0 = -__builtin_inff();
      if (kbase + 16 + laneN > qidx) s1 = -__builtin_inff();
      float rv = fmaxf(s0, s1);
      for (int off = 1; off < 16; off <<= 1) rv = fmaxf(rv, __shfl_xor(rv, off, 32));
      float mnew = fmaxf(mrow[r], rv);
      float p0 = __expf(s0 - mnew), p1 = __expf(s1 - mnew);
      float rs = p0 + p1;
      for (int off = 1; off < 16; off <<= 1) rs += __shfl_xor(rs, off, 32);
      float alpha = __expf(mrow[r] - mnew);
      lrow[r] = lrow[r] * alpha + rs;
      mrow[r] = mnew;
#pragma unroll
      for (int t = 0; t < 4; ++t) o[t][r] *= alpha;
      pp[(hi * 8 + r) * 40 + laneN]      = f32_to_bf16(p0);
      pp[(hi * 8 + r) * 40 + 16 + laneN] = f32_to_bf16(p1);
    }

    // re-fragment P as A-matrix (wave-private LDS; same-wave DS ops in order)
    FragB16 pf, vf[4];
#pragma unroll
    for (int v = 0; v < 8; ++v)
      pf.u[v] = ldsP[wave * 320 + laneN * 20 + a_dw(v, hi)];
#pragma unroll
    for (int t = 0; t < 4; ++t)
#pragma unroll
      for (int v = 0; v < 8; ++v)
        vf[t].u[v] = curV[(t * 16 + laneN) * 20 + hi * 8 + v];
#pragma unroll
    for (int t = 0; t < 4; ++t)
      o[t] = WMMA_BF16(pf.v, vf[t].v, o[t]);

    // finish staging next V tile (transpose), sync
    if (more) {
      unsigned short* vs = (unsigned short*)nxtV;
      const unsigned short* ds = (const unsigned short*)&vreg;
#pragma unroll
      for (int j = 0; j < 8; ++j) vs[(cg * 8 + j) * 40 + row] = ds[j];
      wait_async();
    }
    __syncthreads();
    buf ^= 1;
  }

  // normalize + store y[b, t, h*64+d] as bf16
#pragma unroll
  for (int t = 0; t < 4; ++t)
#pragma unroll
    for (int r = 0; r < 8; ++r) {
      int qrow = q0 + wave * 16 + hi * 8 + r;
      float val = o[t][r] / lrow[r];
      Yb[(size_t)(b * 2048 + qrow) * 1024 + h * 64 + t * 16 + laneN] = f32_to_bf16(val);
    }
}

// ---------------------------------------------------------------- launch
extern "C" void kernel_launch(void* const* d_in, const int* in_sizes, int n_in,
                              void* d_out, int out_size, void* d_ws, size_t ws_size,
                              hipStream_t stream) {
  const float* x      = (const float*)d_in[0];   // [4,2048,1024]
  const float* W_attn = (const float*)d_in[1];   // [1024,3072]
  const float* b_attn = (const float*)d_in[2];   // [3072]
  const float* W_proj = (const float*)d_in[3];   // [1024,1024]
  const float* b_proj = (const float*)d_in[4];   // [1024]
  float* out = (float*)d_out;

  char* ws = (char*)d_ws;
  const size_t MB = 1024 * 1024;
  unsigned short* xb  = (unsigned short*)(ws + 0);         // 16MB (reused as yb)
  unsigned short* wab = (unsigned short*)(ws + 16 * MB);   // 6MB
  unsigned short* wpb = (unsigned short*)(ws + 22 * MB);   // 2MB
  unsigned short* qb  = (unsigned short*)(ws + 24 * MB);   // 16MB
  unsigned short* kb  = (unsigned short*)(ws + 40 * MB);   // 16MB
  unsigned short* vb  = (unsigned short*)(ws + 56 * MB);   // 16MB
  unsigned short* yb  = xb;  // x dead after GEMM1; stream-ordered reuse

  cvt_f32_bf16<<<8192, 256, 0, stream>>>(x, xb, 8388608);
  cvt_f32_bf16<<<3072, 256, 0, stream>>>(W_attn, wab, 3145728);
  cvt_f32_bf16<<<1024, 256, 0, stream>>>(W_proj, wpb, 1048576);

  gemm_qkv<<<64 * 24, 256, 0, stream>>>(xb, wab, b_attn, qb, kb, vb);
  flash_attn<<<1024, 256, 0, stream>>>(qb, kb, vb, yb);
  gemm_proj<<<64 * 8, 256, 0, stream>>>(yb, wpb, b_proj, out);
}
